// GENPYG_14087492730938
// MI455X (gfx1250) — matrix operand order repaired and checked by reading, not compiled, not run on hardware
//
#include <hip/hip_runtime.h>

// ---------------------------------------------------------------------------
// Problem constants (from reference setup_inputs)
// ---------------------------------------------------------------------------
#define B   32
#define P   2048
#define NN  1024
#define EE  8192
#define DH  128
#define STEPS 4

typedef __attribute__((ext_vector_type(16))) __bf16 v16bf;
typedef __attribute__((ext_vector_type(8)))  float  v8f;
typedef __attribute__((ext_vector_type(4)))  unsigned int u32x4;

// ---------------------------------------------------------------------------
// WMMA helpers: v_wmma_f32_16x16x32_bf16, D(16x16,f32) = A(16x32) B(32x16) + C
// Storage convention: "K-contiguous rows" (A tile [M][ldk], B tile [N][ldk]).
// Per-lane layouts (CDNA5 ISA 7.12.2):
//   A: lane L -> row m0+(L&15); elems 0..7  = K k0+8*(L>>4)   .. +7
//                               elems 8..15 = K k0+16+8*(L>>4).. +7
//   B: lane L -> col n0+(L&15); same K order as A
//   C/D: elem v -> row m0+v+8*(L>>4), col n0+(L&15)
// ---------------------------------------------------------------------------
__device__ __forceinline__ v16bf load_a_frag(const __bf16* tile, int ldk,
                                             int m0, int k0, int lane) {
  union { v16bf v; u32x4 q[2]; } f;
  const __bf16* p = tile + (m0 + (lane & 15)) * ldk + k0 + ((lane >> 4) << 3);
  f.q[0] = *(const u32x4*)p;
  f.q[1] = *(const u32x4*)(p + 16);
  return f.v;
}

__device__ __forceinline__ v16bf load_b_frag(const __bf16* tileT, int ldk,
                                             int n0, int k0, int lane) {
  union { v16bf v; u32x4 q[2]; } f;
  const __bf16* p = tileT + (n0 + (lane & 15)) * ldk + k0 + ((lane >> 4) << 3);
  f.q[0] = *(const u32x4*)p;
  f.q[1] = *(const u32x4*)(p + 16);
  return f.v;
}

__device__ __forceinline__ v8f wmma_bf16(v16bf a, v16bf b, v8f c) {
  return __builtin_amdgcn_wmma_f32_16x16x32_bf16(false, a, false, b,
                                                 (short)0, c, false, false);
}

// ---------------------------------------------------------------------------
// CDNA5 async global->LDS DMA (ASYNCcnt path).  The low 32 bits of a flat
// shared-aperture pointer are the wave-relative LDS byte address (ISA 10.2),
// which is what GLOBAL_LOAD_ASYNC_TO_LDS expects in VDST.
// ---------------------------------------------------------------------------
__device__ __forceinline__ void async_ld_b128(const __bf16* gsrc, __bf16* ldst) {
  unsigned int l = (unsigned int)(unsigned long long)ldst;
  asm volatile("global_load_async_to_lds_b128 %0, %1, off"
               :: "v"(l), "v"(gsrc) : "memory");
}
__device__ __forceinline__ void async_wait0() {
  asm volatile("s_wait_asynccnt 0x0" ::: "memory");
}

// Padded K-strides (multiples of 8; keeps 16B alignment, shifts LDS banks)
#define KS  136   // for 128-wide K tiles
#define TKS 72    // for 64-wide K tiles

// ===========================================================================
// Graph preprocessing
// ===========================================================================
__global__ void deg_init_kernel(float* deg) {
  int n = blockIdx.x * 256 + threadIdx.x;
  if (n < NN) deg[n] = 1.0f;                    // self-loop contribution
}

__global__ void deg_edge_kernel(const int* edge, float* deg) {
  int e = blockIdx.x * 256 + threadIdx.x;
  if (e < EE) atomicAdd(&deg[edge[EE + e]], 1.0f);   // col = edge[1][e]
}

__global__ void dinv_kernel(const float* deg, float* dinv) {
  int n = blockIdx.x * 256 + threadIdx.x;
  if (n < NN) dinv[n] = rsqrtf(deg[n]);         // deg >= 1 always
}

__global__ void norm_kernel(const int* edge, const float* dinv, float* nrm) {
  int e = blockIdx.x * 256 + threadIdx.x;
  if (e < EE)            nrm[e] = dinv[edge[e]] * dinv[edge[EE + e]];
  else if (e < EE + NN)  { int n = e - EE; nrm[e] = dinv[n] * dinv[n]; }
}

// ===========================================================================
// Weight preprocessing: bf16 "B-layout" transposes out[j*128+k] = in[k*128+j]
// ===========================================================================
__global__ void prep_weights_kernel(const float* eW1, const float* eW2,
                                    const float* gW,  const float* dW0,
                                    const float* dW1,
                                    __bf16* We2, __bf16* We3, __bf16* Wg,
                                    __bf16* Wd1, __bf16* Wd2) {
  int i = blockIdx.x * 256 + threadIdx.x;       // 5 * 16384
  int m = i >> 14, r = i & 16383;
  int j = r >> 7, k = r & 127;
  switch (m) {
    case 0: We2[r] = (__bf16)eW1[k * 128 + j]; break;
    case 1: We3[r] = (__bf16)eW2[k * 128 + j]; break;
    case 2: Wg [r] = (__bf16)gW [k * 128 + j]; break;   // rows 0..127 of gcn_w
    case 3: Wd1[r] = (__bf16)dW0[k * 128 + j]; break;   // rows 0..127 of dec W0
    case 4: Wd2[r] = (__bf16)dW1[k * 128 + j]; break;
  }
}

// posW[n][h] = pos[n,0]*gcn_w[128][h] + pos[n,1]*gcn_w[129][h] (batch-invariant)
__global__ void posw_kernel(const float* pos, const float* gW, float* posW) {
  int i = blockIdx.x * 256 + threadIdx.x;       // NN*128
  int n = i >> 7, h = i & 127;
  posW[i] = pos[n * 2] * gW[128 * 128 + h] + pos[n * 2 + 1] * gW[129 * 128 + h];
}

// ===========================================================================
// Softmax denominators: rZ[b,p] = 1 / sum_n exp(-||x_bp - pos_n||^2)
// ===========================================================================
__global__ __launch_bounds__(256) void zden_kernel(const float* x,
                                                   const float* pos, float* rZ) {
  __shared__ float sp[NN * 2];
  for (int i = threadIdx.x; i < NN * 2; i += 256) sp[i] = pos[i];
  __syncthreads();
  int idx = blockIdx.x * 256 + threadIdx.x;     // b*P + p
  float x0 = x[idx * 2], x1 = x[idx * 2 + 1];
  float z = 0.0f;
  for (int n = 0; n < NN; ++n) {
    float d0 = x0 - sp[n * 2], d1 = x1 - sp[n * 2 + 1];
    z += __expf(-(d0 * d0 + d1 * d1));
  }
  rZ[idx] = 1.0f / z;
}

// ===========================================================================
// Encoder MLP: (xc||yc)[5] -> 128 (VALU) -> relu -> 128 (WMMA) -> relu
//              -> 128 (WMMA); writes embT[b][h][p] bf16 (K-contiguous in p)
// ===========================================================================
__global__ __launch_bounds__(256) void encoder_kernel(
    const float* xc, const float* yc, const float* W0, const float* b0,
    const float* b1, const float* b2, const __bf16* We2, const __bf16* We3,
    __bf16* embT) {
  __shared__ __align__(16) __bf16 sH[128 * KS];
  __shared__ float sXY[128 * 5];
  int b = blockIdx.y, p0 = blockIdx.x * 128;
  int tid = threadIdx.x, lane = tid & 31, w = tid >> 5;

  for (int i = tid; i < 128 * 5; i += 256) {
    int r = i / 5, c = i % 5, p = p0 + r;
    sXY[i] = (c < 2) ? xc[((size_t)b * P + p) * 2 + c]
                     : yc[((size_t)b * P + p) * 3 + (c - 2)];
  }
  __syncthreads();
  // layer 1 (K=5, VALU)
  for (int i = tid; i < 128 * 128; i += 256) {
    int r = i >> 7, j = i & 127;
    float acc = b0[j];
#pragma unroll
    for (int k = 0; k < 5; ++k) acc += sXY[r * 5 + k] * W0[k * 128 + j];
    sH[r * KS + j] = (__bf16)fmaxf(acc, 0.0f);
  }
  __syncthreads();

  int m0 = w * 16;
  // layer 2 (WMMA, B frags straight from global We2)
  v8f c[8] = {};
  for (int kk = 0; kk < 128; kk += 32) {
    v16bf a = load_a_frag(sH, KS, m0, kk, lane);
#pragma unroll
    for (int t = 0; t < 8; ++t)
      c[t] = wmma_bf16(a, load_b_frag(We2, 128, t * 16, kk, lane), c[t]);
  }
  // bias + relu, in-place (wave only touches its own 16 rows)
#pragma unroll
  for (int t = 0; t < 8; ++t) {
    int n = t * 16 + (lane & 15);
    float bn = b1[n];
#pragma unroll
    for (int v = 0; v < 8; ++v) {
      int m = m0 + v + ((lane >> 4) << 3);
      sH[m * KS + n] = (__bf16)fmaxf(c[t][v] + bn, 0.0f);
    }
  }
  __syncthreads();
  // layer 3 (WMMA), no relu, write transposed embT[b][h][p]
  v8f d[8] = {};
  for (int kk = 0; kk < 128; kk += 32) {
    v16bf a = load_a_frag(sH, KS, m0, kk, lane);
#pragma unroll
    for (int t = 0; t < 8; ++t)
      d[t] = wmma_bf16(a, load_b_frag(We3, 128, t * 16, kk, lane), d[t]);
  }
#pragma unroll
  for (int t = 0; t < 8; ++t) {
    int h = t * 16 + (lane & 15);
    float bn = b2[h];
#pragma unroll
    for (int v = 0; v < 8; ++v) {
      int m = m0 + v + ((lane >> 4) << 3);
      embT[((size_t)b * DH + h) * P + p0 + m] = (__bf16)(d[t][v] + bn);
    }
  }
}

// ===========================================================================
// lat = Sc^T @ Emb  (per batch: M=1024 nodes, K=P=2048, Nh=128)
// Score tiles built on the fly (exp * rZ) into LDS A-layout; Emb^T streamed
// with double-buffered async global->LDS DMA overlapped with exp + WMMA.
// ===========================================================================
__global__ __launch_bounds__(256) void scores_lat_kernel(
    const float* xc, const float* pos, const float* rZ, const __bf16* embT,
    __bf16* latbf) {
  __shared__ __align__(16) __bf16 sSc[64 * TKS];      // [node][point]
  __shared__ __align__(16) __bf16 sE[2][128 * TKS];   // [h][point] x2
  __shared__ float sPos[128], sX[2][128], sRZ[2][64];
  int b = blockIdx.y, n0 = blockIdx.x * 64;
  int tid = threadIdx.x, lane = tid & 31, w = tid >> 5;
  for (int i = tid; i < 128; i += 256) sPos[i] = pos[n0 * 2 + i];

  // prologue: async-stage emb tile 0 + point data 0
  for (int i = tid; i < 128 * 8; i += 256) {
    int h = i >> 3, ch = (i & 7) * 8;
    async_ld_b128(&embT[((size_t)b * DH + h) * P + ch], &sE[0][h * TKS + ch]);
  }
  if (tid < 64) {
    sX[0][tid * 2]     = xc[((size_t)b * P + tid) * 2];
    sX[0][tid * 2 + 1] = xc[((size_t)b * P + tid) * 2 + 1];
    sRZ[0][tid]        = rZ[b * P + tid];
  }

  v8f c[4] = {};
  int m0 = (w & 3) * 16, nh = (w >> 2) * 64;
  for (int kc = 0; kc < P; kc += 64) {
    int cur = (kc >> 6) & 1, nxt = cur ^ 1;
    async_wait0();
    __syncthreads();   // sE[cur] resident; prior WMMA reads of sE[nxt]/sSc done
    if (kc + 64 < P) {
      for (int i = tid; i < 128 * 8; i += 256) {     // prefetch next tile
        int h = i >> 3, ch = (i & 7) * 8;
        async_ld_b128(&embT[((size_t)b * DH + h) * P + kc + 64 + ch],
                      &sE[nxt][h * TKS + ch]);
      }
      if (tid < 64) {
        int p = kc + 64 + tid;
        sX[nxt][tid * 2]     = xc[((size_t)b * P + p) * 2];
        sX[nxt][tid * 2 + 1] = xc[((size_t)b * P + p) * 2 + 1];
        sRZ[nxt][tid]        = rZ[b * P + p];
      }
    }
    for (int i = tid; i < 64 * 64; i += 256) {       // score tile (exp-bound)
      int nl = i >> 6, pp = i & 63;
      float d0 = sX[cur][pp * 2] - sPos[nl * 2];
      float d1 = sX[cur][pp * 2 + 1] - sPos[nl * 2 + 1];
      sSc[nl * TKS + pp] =
          (__bf16)(__expf(-(d0 * d0 + d1 * d1)) * sRZ[cur][pp]);
    }
    __syncthreads();
#pragma unroll
    for (int ks = 0; ks < 64; ks += 32) {
      v16bf a = load_a_frag(sSc, TKS, m0, ks, lane);
#pragma unroll
      for (int t = 0; t < 4; ++t)
        c[t] = wmma_bf16(a, load_b_frag(sE[cur], TKS, nh + t * 16, ks, lane),
                         c[t]);
    }
  }
#pragma unroll
  for (int t = 0; t < 4; ++t) {
    int h = nh + t * 16 + (lane & 15);
#pragma unroll
    for (int v = 0; v < 8; ++v) {
      int n = n0 + m0 + v + ((lane >> 4) << 3);
      latbf[((size_t)b * NN + n) * DH + h] = (__bf16)c[t][v];
    }
  }
}

// ===========================================================================
// GCN feature GEMM: H = lat @ Wg(128x128) + posW  (per batch M=1024,K=128)
// A tile staged with async global->LDS DMA.
// ===========================================================================
__global__ __launch_bounds__(256) void gcn_gemm_kernel(
    const __bf16* latbf, const __bf16* Wg, const float* posW, float* H) {
  __shared__ __align__(16) __bf16 sA[128 * KS];
  int b = blockIdx.y, n0 = blockIdx.x * 128;
  int tid = threadIdx.x, lane = tid & 31, w = tid >> 5;
  for (int i = tid; i < 128 * 16; i += 256) {
    int r = i >> 4, ch = (i & 15) * 8;
    async_ld_b128(&latbf[((size_t)b * NN + n0 + r) * DH + ch],
                  &sA[r * KS + ch]);
  }
  async_wait0();
  __syncthreads();
  v8f c[8] = {};
  int m0 = w * 16;
  for (int kk = 0; kk < 128; kk += 32) {
    v16bf a = load_a_frag(sA, KS, m0, kk, lane);
#pragma unroll
    for (int t = 0; t < 8; ++t)
      c[t] = wmma_bf16(a, load_b_frag(Wg, 128, t * 16, kk, lane), c[t]);
  }
#pragma unroll
  for (int t = 0; t < 8; ++t) {
    int h = t * 16 + (lane & 15);
#pragma unroll
    for (int v = 0; v < 8; ++v) {
      int n = n0 + m0 + v + ((lane >> 4) << 3);
      H[((size_t)b * NN + n) * DH + h] = c[t][v] + posW[n * DH + h];
    }
  }
}

// Edge scatter: agg[b,col,:] += norm_e * H[b,row,:]  (128 thr per edge inst)
__global__ __launch_bounds__(256) void scatter_kernel(
    const float* H, const int* edge, const float* nrm, float* agg) {
  int g = blockIdx.x * 2 + (threadIdx.x >> 7);
  if (g >= B * (EE + NN)) return;
  int h = threadIdx.x & 127;
  int b = g / (EE + NN), e = g - b * (EE + NN);
  int r, c;
  if (e < EE) { r = edge[e]; c = edge[EE + e]; }
  else        { r = c = e - EE; }
  float val = nrm[e] * H[((size_t)b * NN + r) * DH + h];
  atomicAdd(&agg[((size_t)b * NN + c) * DH + h], val);
}

// lat = agg + bias; emit row-major bf16 (next GEMM A) + transposed (decode B)
__global__ __launch_bounds__(256) void gcn_finish_kernel(
    const float* agg, const float* bias, __bf16* latbf, __bf16* latT) {
  int i = blockIdx.x * 256 + threadIdx.x;       // B*NN*128
  int h = i & 127, n = (i >> 7) & (NN - 1), b = i >> 17;
  float v = agg[i] + bias[h];
  latbf[i] = (__bf16)v;
  latT[((size_t)b * DH + h) * NN + n] = (__bf16)v;
}

// ===========================================================================
// z = St @ lat  (per batch: M=P=2048, K=N=1024, Nh=128); St built on the fly,
// lat^T streamed with double-buffered async DMA.
// ===========================================================================
__global__ __launch_bounds__(256) void decode_gemm_kernel(
    const float* xt, const float* pos, const float* rZt, const __bf16* latT,
    float* zout) {
  __shared__ __align__(16) __bf16 sSt[64 * TKS];      // [point][node]
  __shared__ __align__(16) __bf16 sL[2][128 * TKS];   // [h][node] x2
  __shared__ float sX[128], sRZ[64], sPos[2][128];
  int b = blockIdx.y, p0 = blockIdx.x * 64;
  int tid = threadIdx.x, lane = tid & 31, w = tid >> 5;
  if (tid < 64) {
    int p = p0 + tid;
    sX[tid * 2]     = xt[((size_t)b * P + p) * 2];
    sX[tid * 2 + 1] = xt[((size_t)b * P + p) * 2 + 1];
    sRZ[tid]        = rZt[b * P + p];
  }
  for (int i = tid; i < 128; i += 256) sPos[0][i] = pos[i];
  for (int i = tid; i < 128 * 8; i += 256) {
    int h = i >> 3, ch = (i & 7) * 8;
    async_ld_b128(&latT[((size_t)b * DH + h) * NN + ch], &sL[0][h * TKS + ch]);
  }
  v8f c[4] = {};
  int m0 = (w & 3) * 16, nh = (w >> 2) * 64;
  for (int kc = 0; kc < NN; kc += 64) {
    int cur = (kc >> 6) & 1, nxt = cur ^ 1;
    async_wait0();
    __syncthreads();
    if (kc + 64 < NN) {
      for (int i = tid; i < 128 * 8; i += 256) {
        int h = i >> 3, ch = (i & 7) * 8;
        async_ld_b128(&latT[((size_t)b * DH + h) * NN + kc + 64 + ch],
                      &sL[nxt][h * TKS + ch]);
      }
      for (int i = tid; i < 128; i += 256) sPos[nxt][i] = pos[(kc + 64) * 2 + i];
    }
    for (int i = tid; i < 64 * 64; i += 256) {
      int pl = i >> 6, nn = i & 63;
      float d0 = sX[pl * 2] - sPos[cur][nn * 2];
      float d1 = sX[pl * 2 + 1] - sPos[cur][nn * 2 + 1];
      sSt[pl * TKS + nn] =
          (__bf16)(__expf(-(d0 * d0 + d1 * d1)) * sRZ[pl]);
    }
    __syncthreads();
#pragma unroll
    for (int ks = 0; ks < 64; ks += 32) {
      v16bf a = load_a_frag(sSt, TKS, m0, ks, lane);
#pragma unroll
      for (int t = 0; t < 4; ++t)
        c[t] = wmma_bf16(a, load_b_frag(sL[cur], TKS, nh + t * 16, ks, lane),
                         c[t]);
    }
  }
#pragma unroll
  for (int t = 0; t < 4; ++t) {
    int h = nh + t * 16 + (lane & 15);
#pragma unroll
    for (int v = 0; v < 8; ++v) {
      int p = p0 + m0 + v + ((lane >> 4) << 3);
      zout[((size_t)b * P + p) * DH + h] = c[t][v];
    }
  }
}

// ===========================================================================
// Decoder MLP: relu(z@Wd1 + xt@Wd1_bot + b) -> relu(@Wd2 + b) -> @W3(128x3)+b
// ===========================================================================
__global__ __launch_bounds__(256) void decoder_kernel(
    const float* zin, const float* xt, const __bf16* Wd1, const float* dW0,
    const float* b1, const __bf16* Wd2, const float* b2, const float* W3,
    const float* b3, float* out) {
  __shared__ __align__(16) __bf16 sH[128 * KS];
  __shared__ float sXT[128 * 2];
  int b = blockIdx.y, p0 = blockIdx.x * 128;
  int tid = threadIdx.x, lane = tid & 31, w = tid >> 5;
  for (int i = tid; i < 256; i += 256)
    sXT[i] = xt[((size_t)b * P + p0) * 2 + i];
  for (int i = tid; i < 128 * 128; i += 256) {
    int r = i >> 7, k = i & 127;
    sH[r * KS + k] = (__bf16)zin[((size_t)b * P + p0 + r) * DH + k];
  }
  __syncthreads();
  int m0 = w * 16;
  // layer 1
  v8f c[8] = {};
  for (int kk = 0; kk < 128; kk += 32) {
    v16bf a = load_a_frag(sH, KS, m0, kk, lane);
#pragma unroll
    for (int t = 0; t < 8; ++t)
      c[t] = wmma_bf16(a, load_b_frag(Wd1, 128, t * 16, kk, lane), c[t]);
  }
#pragma unroll
  for (int t = 0; t < 8; ++t) {
    int j = t * 16 + (lane & 15);
    float wb0 = dW0[128 * 128 + j], wb1 = dW0[129 * 128 + j], bj = b1[j];
#pragma unroll
    for (int v = 0; v < 8; ++v) {
      int m = m0 + v + ((lane >> 4) << 3);
      float val = c[t][v] + sXT[m * 2] * wb0 + sXT[m * 2 + 1] * wb1 + bj;
      sH[m * KS + j] = (__bf16)fmaxf(val, 0.0f);   // in-place, own rows
    }
  }
  // layer 2
  v8f d[8] = {};
  for (int kk = 0; kk < 128; kk += 32) {
    v16bf a = load_a_frag(sH, KS, m0, kk, lane);
#pragma unroll
    for (int t = 0; t < 8; ++t)
      d[t] = wmma_bf16(a, load_b_frag(Wd2, 128, t * 16, kk, lane), d[t]);
  }
#pragma unroll
  for (int t = 0; t < 8; ++t) {
    int j = t * 16 + (lane & 15);
    float bj = b2[j];
#pragma unroll
    for (int v = 0; v < 8; ++v) {
      int m = m0 + v + ((lane >> 4) << 3);
      sH[m * KS + j] = (__bf16)fmaxf(d[t][v] + bj, 0.0f);
    }
  }
  __syncthreads();
  // layer 3: 128 -> 3 (VALU)
  if (tid < 128) {
    float a0 = b3[0], a1 = b3[1], a2 = b3[2];
    for (int k = 0; k < 128; ++k) {
      float hv = (float)sH[tid * KS + k];
      a0 += hv * W3[k * 3 + 0];
      a1 += hv * W3[k * 3 + 1];
      a2 += hv * W3[k * 3 + 2];
    }
    float* o = out + ((size_t)b * P + p0 + tid) * 3;
    o[0] = a0; o[1] = a1; o[2] = a2;
  }
}

// ===========================================================================
// Host launcher
// ===========================================================================
extern "C" void kernel_launch(void* const* d_in, const int* in_sizes, int n_in,
                              void* d_out, int out_size, void* d_ws,
                              size_t ws_size, hipStream_t stream) {
  const float* xc  = (const float*)d_in[0];
  const float* yc  = (const float*)d_in[1];
  const float* xt  = (const float*)d_in[2];
  const float* pos = (const float*)d_in[3];
  const int*   edg = (const int*)d_in[4];
  const float* eW0 = (const float*)d_in[5];
  const float* eB0 = (const float*)d_in[6];
  const float* eW1 = (const float*)d_in[7];
  const float* eB1 = (const float*)d_in[8];
  const float* eW2 = (const float*)d_in[9];
  const float* eB2 = (const float*)d_in[10];
  const float* dW0 = (const float*)d_in[11];
  const float* dB0 = (const float*)d_in[12];
  const float* dW1 = (const float*)d_in[13];
  const float* dB1 = (const float*)d_in[14];
  const float* dW2 = (const float*)d_in[15];
  const float* dB2 = (const float*)d_in[16];
  const float* gW  = (const float*)d_in[17];
  const float* gB  = (const float*)d_in[18];

  char* wp = (char*)d_ws;
  auto carve = [&](size_t bytes) -> void* {
    void* r = (void*)wp;
    wp += (bytes + 255) & ~(size_t)255;
    return r;
  };
  float*  deg   = (float*)carve(NN * 4);
  float*  dinv  = (float*)carve(NN * 4);
  float*  nrm   = (float*)carve((size_t)(EE + NN) * 4);
  float*  rZc   = (float*)carve((size_t)B * P * 4);
  float*  rZt   = (float*)carve((size_t)B * P * 4);
  float*  posW  = (float*)carve((size_t)NN * DH * 4);
  __bf16* We2   = (__bf16*)carve(16384 * 2);
  __bf16* We3   = (__bf16*)carve(16384 * 2);
  __bf16* Wg    = (__bf16*)carve(16384 * 2);
  __bf16* Wd1   = (__bf16*)carve(16384 * 2);
  __bf16* Wd2   = (__bf16*)carve(16384 * 2);
  __bf16* embT  = (__bf16*)carve((size_t)B * DH * P * 2);
  __bf16* latbf = (__bf16*)carve((size_t)B * NN * DH * 2);
  __bf16* latT  = (__bf16*)carve((size_t)B * NN * DH * 2);
  float*  H     = (float*)carve((size_t)B * NN * DH * 4);
  float*  agg   = (float*)carve((size_t)B * NN * DH * 4);
  float*  z     = (float*)carve((size_t)B * P * DH * 4);

  // graph prep
  deg_init_kernel<<<(NN + 255) / 256, 256, 0, stream>>>(deg);
  deg_edge_kernel<<<(EE + 255) / 256, 256, 0, stream>>>(edg, deg);
  dinv_kernel<<<(NN + 255) / 256, 256, 0, stream>>>(deg, dinv);
  norm_kernel<<<(EE + NN + 255) / 256, 256, 0, stream>>>(edg, dinv, nrm);

  // weights / pos term
  prep_weights_kernel<<<5 * 16384 / 256, 256, 0, stream>>>(
      eW1, eW2, gW, dW0, dW1, We2, We3, Wg, Wd1, Wd2);
  posw_kernel<<<NN * DH / 256, 256, 0, stream>>>(pos, gW, posW);

  // encode
  encoder_kernel<<<dim3(P / 128, B), 256, 0, stream>>>(
      xc, yc, eW0, eB0, eB1, eB2, We2, We3, embT);
  zden_kernel<<<B * P / 256, 256, 0, stream>>>(xc, pos, rZc);
  zden_kernel<<<B * P / 256, 256, 0, stream>>>(xt, pos, rZt);
  scores_lat_kernel<<<dim3(NN / 64, B), 256, 0, stream>>>(
      xc, pos, rZc, embT, latbf);

  // message passing
  for (int s = 0; s < STEPS; ++s) {
    hipMemsetAsync(agg, 0, (size_t)B * NN * DH * 4, stream);
    gcn_gemm_kernel<<<dim3(NN / 128, B), 256, 0, stream>>>(latbf, Wg, posW, H);
    scatter_kernel<<<B * (EE + NN) / 2, 256, 0, stream>>>(H, edg, nrm, agg);
    gcn_finish_kernel<<<B * NN * DH / 256, 256, 0, stream>>>(agg, gB, latbf,
                                                             latT);
  }

  // decode
  decode_gemm_kernel<<<dim3(P / 64, B), 256, 0, stream>>>(xt, pos, rZt, latT,
                                                          z);
  decoder_kernel<<<dim3(P / 128, B), 256, 0, stream>>>(
      z, xt, Wd1, dW0, dB0, Wd2, dB1, dW2, dB2, (float*)d_out);
}
